// FactorAtt_ConvRelPosEnc_79680233275828
// MI455X (gfx1250) — matrix operand order repaired and checked by reading, not compiled, not run on hardware
//
#include <hip/hip_runtime.h>
#include <hip/hip_bf16.h>

typedef __attribute__((ext_vector_type(16))) __bf16 v16bf;
typedef __attribute__((ext_vector_type(8)))  float  v8f;

#define B_   8
#define N_   4096
#define C_   256
#define HDS  8
#define CH_  32
#define M_   (B_ * N_)   // 32768
#define IMG  64

// ---- helpers ---------------------------------------------------------------

__device__ __forceinline__ unsigned short f2bf(float f) {
  unsigned int u = __float_as_uint(f);
  u += 0x7FFFu + ((u >> 16) & 1u);            // round-to-nearest-even
  return (unsigned short)(u >> 16);
}

// A fragment (16x32 bf16, MxK) from row-major bf16 tile, leading dim `ld`.
// Lane m = lane&15; K = {half*8+0..7, 16+half*8+0..7}  (ISA 7.12.2 table)
__device__ __forceinline__ v16bf frag_a_bf16(const unsigned short* tile, int ld) {
  const int lane = threadIdx.x & 31;
  const int half = lane >> 4;
  const int m    = lane & 15;
  const unsigned short* p = tile + (size_t)m * ld + half * 8;
  union { v16bf v; uint4 q[2]; } u;
  u.q[0] = *reinterpret_cast<const uint4*>(p);
  u.q[1] = *reinterpret_cast<const uint4*>(p + 16);
  return u.v;
}

// B fragment (32x16 bf16, KxN) from B^T stored row-major [n][k], leading dim `ld`.
// Lane n = lane&15; K = {half*8+0..7, 16+half*8+0..7}  (matches A fragment)
__device__ __forceinline__ v16bf frag_b_bf16(const unsigned short* tileT, int ld) {
  const int lane = threadIdx.x & 31;
  const int half = lane >> 4;
  const int n    = lane & 15;
  const unsigned short* p = tileT + (size_t)n * ld + half * 8;
  union { v16bf v; uint4 q[2]; } u;
  u.q[0] = *reinterpret_cast<const uint4*>(p);
  u.q[1] = *reinterpret_cast<const uint4*>(p + 16);
  return u.v;
}

// A fragment converted on the fly from row-major f32, leading dim `ld`.
__device__ __forceinline__ v16bf frag_a_from_f32(const float* rowbase, int ld) {
  const int lane = threadIdx.x & 31;
  const int half = lane >> 4;
  const int m    = lane & 15;
  const float* p = rowbase + (size_t)m * ld;
  union { v16bf v; unsigned short e[16]; } u;
#pragma unroll
  for (int j = 0; j < 8; ++j) u.e[j]     = f2bf(p[half * 8 + j]);
#pragma unroll
  for (int j = 0; j < 8; ++j) u.e[8 + j] = f2bf(p[16 + half * 8 + j]);
  return u.v;
}

// Store C/D fragment (16x16 f32) to row-major f32, leading dim `ld`, + bias.
__device__ __forceinline__ void store_c(float* base, int ld, const v8f& c, float bias) {
  const int lane = threadIdx.x & 31;
  const int half = lane >> 4;
  const int n    = lane & 15;
#pragma unroll
  for (int r = 0; r < 8; ++r) base[(size_t)(r + half * 8) * ld + n] = c[r] + bias;
}

// ---- kernels ---------------------------------------------------------------

// Convert weights to bf16, zero softmax + kv accumulators.
__global__ void k_prep(const float* __restrict__ Wqkv, const float* __restrict__ Wproj,
                       unsigned short* __restrict__ wqb, unsigned short* __restrict__ wpb,
                       float* __restrict__ sume, float* __restrict__ kvbuf) {
  int i = blockIdx.x * blockDim.x + threadIdx.x;
  if (i < 3 * C_ * C_) wqb[i] = f2bf(Wqkv[i]);
  if (i < C_ * C_)     wpb[i] = f2bf(Wproj[i]);
  if (i < B_ * C_)     sume[i] = 0.0f;
  if (i < 64 * 1024)   kvbuf[i] = 0.0f;
}

// x -> bf16 (4 elements / thread)
__global__ void k_cvt_x(const float* __restrict__ x, unsigned short* __restrict__ xb) {
  int i = blockIdx.x * blockDim.x + threadIdx.x;
  float4 f = reinterpret_cast<const float4*>(x)[i];
  unsigned int lo = (unsigned int)f2bf(f.x) | ((unsigned int)f2bf(f.y) << 16);
  unsigned int hi = (unsigned int)f2bf(f.z) | ((unsigned int)f2bf(f.w) << 16);
  reinterpret_cast<uint2*>(xb)[i] = make_uint2(lo, hi);
}

// qkv = x @ Wqkv^T  (M=32768, N=768, K=256). Block 256 thr = 8 waves,
// block tile 64(M) x 128(N), wave tile 16 x 64. Software-pipelined K loop.
__global__ __launch_bounds__(256) void k_gemm_qkv(const unsigned short* __restrict__ xb,
                                                  const unsigned short* __restrict__ wb,
                                                  float* __restrict__ qkv) {
  const int wave = threadIdx.x >> 5;
  const int wm = wave & 3, wn = wave >> 2;
  const int m0 = blockIdx.y * 64 + wm * 16;
  const int n0 = blockIdx.x * 128 + wn * 64;
  const unsigned short* abase = xb + (size_t)m0 * C_;
  const unsigned short* wbase = wb + (size_t)n0 * C_;

  v8f acc[4] = {};
  v16bf a  = frag_a_bf16(abase, C_);
  v16bf b0 = frag_b_bf16(wbase + 0 * 16 * C_, C_);
  v16bf b1 = frag_b_bf16(wbase + 1 * 16 * C_, C_);
  v16bf b2 = frag_b_bf16(wbase + 2 * 16 * C_, C_);
  v16bf b3 = frag_b_bf16(wbase + 3 * 16 * C_, C_);
#pragma unroll
  for (int k0 = 0; k0 < C_; k0 += 32) {
    const bool last = (k0 + 32 >= C_);
    v16bf an  = last ? a  : frag_a_bf16(abase + k0 + 32, C_);
    v16bf bn0 = last ? b0 : frag_b_bf16(wbase + 0 * 16 * C_ + k0 + 32, C_);
    v16bf bn1 = last ? b1 : frag_b_bf16(wbase + 1 * 16 * C_ + k0 + 32, C_);
    v16bf bn2 = last ? b2 : frag_b_bf16(wbase + 2 * 16 * C_ + k0 + 32, C_);
    v16bf bn3 = last ? b3 : frag_b_bf16(wbase + 3 * 16 * C_ + k0 + 32, C_);
    acc[0] = __builtin_amdgcn_wmma_f32_16x16x32_bf16(false, a, false, b0, (short)0, acc[0], false, false);
    acc[1] = __builtin_amdgcn_wmma_f32_16x16x32_bf16(false, a, false, b1, (short)0, acc[1], false, false);
    acc[2] = __builtin_amdgcn_wmma_f32_16x16x32_bf16(false, a, false, b2, (short)0, acc[2], false, false);
    acc[3] = __builtin_amdgcn_wmma_f32_16x16x32_bf16(false, a, false, b3, (short)0, acc[3], false, false);
    a = an; b0 = bn0; b1 = bn1; b2 = bn2; b3 = bn3;
  }
#pragma unroll
  for (int t = 0; t < 4; ++t)
    store_c(qkv + (size_t)m0 * 768 + n0 + 16 * t, 768, acc[t], 0.0f);
}

// Per-(b, channel) sum of exp(k) over the sequence axis (k ~ N(0,1): safe w/o max).
__global__ __launch_bounds__(256) void k_softsum(const float* __restrict__ qkv,
                                                 float* __restrict__ sume) {
  const int b = blockIdx.x >> 4;
  const int chunk = blockIdx.x & 15;
  const int c = threadIdx.x;
  float acc = 0.0f;
  int base = b * N_ + chunk * 256;
  for (int i = 0; i < 256; ++i)
    acc += __expf(qkv[(size_t)(base + i) * 768 + C_ + c]);
  atomicAdd(&sume[b * C_ + c], acc);
}

// kv[b,h] += exp(k)^T @ v  over a 512-long K segment (split-K, 8 segments).
// One block / (b,h,seg), 4 waves, each owns a 16x16 quadrant of 32x32 output.
// Row normalization (1/sum, Ch^-0.5) is applied later in k_fa.
__global__ __launch_bounds__(128) void k_kv(const float* __restrict__ qkv,
                                            float* __restrict__ kvbuf) {
  __shared__ unsigned short As[32 * 32];  // As[c1][kk] = bf16(exp(k[n0+kk, cbase+c1]))
  __shared__ unsigned short Bs[32 * 32];  // Bs[c2][kk] = bf16(v[n0+kk, cbase+c2]) (B^T)
  const int bh = blockIdx.x;
  const int b = bh >> 3, h = bh & 7;
  const int cbase = h * CH_;
  const int seg0 = blockIdx.y * 512;
  const int wave = threadIdx.x >> 5;
  const int mq = wave & 1, nq = wave >> 1;
  v8f acc = {};
  for (int n0 = seg0; n0 < seg0 + 512; n0 += 32) {
#pragma unroll
    for (int i = 0; i < 8; ++i) {
      int e = threadIdx.x + i * 128;
      int kk = e >> 5, cc = e & 31;
      size_t row = (size_t)(b * N_ + n0 + kk) * 768;
      As[cc * 32 + kk] = f2bf(__expf(qkv[row + C_ + cbase + cc]));
      Bs[cc * 32 + kk] = f2bf(qkv[row + 2 * C_ + cbase + cc]);
    }
    __syncthreads();
    v16bf a  = frag_a_bf16(As + mq * 16 * 32, 32);
    v16bf bb = frag_b_bf16(Bs + nq * 16 * 32, 32);
    acc = __builtin_amdgcn_wmma_f32_16x16x32_bf16(false, a, false, bb,
                                                  (short)0, acc, false, false);
    __syncthreads();
  }
  const int lane = threadIdx.x & 31;
  const int half = lane >> 4, n = lane & 15;
#pragma unroll
  for (int r = 0; r < 8; ++r) {
    int c1 = mq * 16 + r + half * 8;
    int c2 = nq * 16 + n;
    atomicAdd(&kvbuf[(size_t)bh * 1024 + c1 * 32 + c2], acc[r]);
  }
}

// factor_att = q @ (kv * diag(scale/sum)). Block 256 thr (8 waves x 16 rows),
// grid (64 bh, 32 row-tiles). kv normalized+scaled while staging transposed in LDS.
__global__ __launch_bounds__(256) void k_fa(const float* __restrict__ qkv,
                                            const float* __restrict__ kvbuf,
                                            const float* __restrict__ sume,
                                            float* __restrict__ fa) {
  __shared__ unsigned short kvT[32 * 32];  // kvT[c2][c1]
  const int bh = blockIdx.x;
  const int b = bh >> 3, h = bh & 7;
  const float sc = 0.17677669529663687f;   // 32^-0.5
#pragma unroll
  for (int i = 0; i < 4; ++i) {
    int e = threadIdx.x + i * 256;
    int c1 = e >> 5, c2 = e & 31;
    float norm = sc / sume[b * C_ + h * CH_ + c1];
    kvT[c2 * 32 + c1] = f2bf(kvbuf[(size_t)bh * 1024 + e] * norm);
  }
  __syncthreads();
  const int wave = threadIdx.x >> 5;
  const int row0 = blockIdx.y * 128 + wave * 16;
  const float* qbase = qkv + (size_t)(b * N_ + row0) * 768 + h * CH_;
  v16bf a  = frag_a_from_f32(qbase, 768);
  v16bf b0 = frag_b_bf16(kvT, 32);
  v16bf b1 = frag_b_bf16(kvT + 16 * 32, 32);
  v8f c0 = {}, c1 = {};
  c0 = __builtin_amdgcn_wmma_f32_16x16x32_bf16(false, a, false, b0, (short)0, c0, false, false);
  c1 = __builtin_amdgcn_wmma_f32_16x16x32_bf16(false, a, false, b1, (short)0, c1, false, false);
  float* obase = fa + (size_t)(b * N_ + row0) * C_ + h * CH_;
  store_c(obase, C_, c0, 0.0f);
  store_c(obase + 16, C_, c1, 0.0f);
}

// Depthwise 3x3 conv on v (as 64x64 image) + bias, gate by q, add factor_att,
// pack result to bf16 for the projection GEMM.
__global__ __launch_bounds__(256) void k_crpe(const float* __restrict__ qkv,
                                              const float* __restrict__ fa,
                                              const float* __restrict__ ker,
                                              const float* __restrict__ kbias,
                                              unsigned short* __restrict__ outp) {
  const int m = blockIdx.x;       // b*4096 + n
  const int c = threadIdx.x;      // channel
  const int b = m >> 12, n = m & 4095;
  const int y = n >> 6, x = n & 63;
  float conv = kbias[c];
  const float* vbase = qkv + 2 * C_ + c;
#pragma unroll
  for (int dy = 0; dy < 3; ++dy) {
    int yy = y + dy - 1;
    if (yy < 0 || yy >= IMG) continue;
#pragma unroll
    for (int dx = 0; dx < 3; ++dx) {
      int xx = x + dx - 1;
      if (xx < 0 || xx >= IMG) continue;
      conv += ker[c * 9 + dy * 3 + dx] *
              vbase[(size_t)(b * N_ + yy * IMG + xx) * 768];
    }
  }
  float qv  = qkv[(size_t)m * 768 + c];
  float val = fa[(size_t)m * C_ + c] + qv * conv;
  outp[(size_t)m * C_ + c] = f2bf(val);
}

// out = pre @ Wproj^T + bproj  (M=32768, N=256, K=256). Software-pipelined.
__global__ __launch_bounds__(256) void k_gemm_proj(const unsigned short* __restrict__ ab,
                                                   const unsigned short* __restrict__ wb,
                                                   const float* __restrict__ bias,
                                                   float* __restrict__ out) {
  const int wave = threadIdx.x >> 5;
  const int wm = wave & 3, wn = wave >> 2;
  const int m0 = blockIdx.y * 64 + wm * 16;
  const int n0 = blockIdx.x * 128 + wn * 64;
  const unsigned short* abase = ab + (size_t)m0 * C_;
  const unsigned short* wbase = wb + (size_t)n0 * C_;

  v8f acc[4] = {};
  v16bf a  = frag_a_bf16(abase, C_);
  v16bf b0 = frag_b_bf16(wbase + 0 * 16 * C_, C_);
  v16bf b1 = frag_b_bf16(wbase + 1 * 16 * C_, C_);
  v16bf b2 = frag_b_bf16(wbase + 2 * 16 * C_, C_);
  v16bf b3 = frag_b_bf16(wbase + 3 * 16 * C_, C_);
#pragma unroll
  for (int k0 = 0; k0 < C_; k0 += 32) {
    const bool last = (k0 + 32 >= C_);
    v16bf an  = last ? a  : frag_a_bf16(abase + k0 + 32, C_);
    v16bf bn0 = last ? b0 : frag_b_bf16(wbase + 0 * 16 * C_ + k0 + 32, C_);
    v16bf bn1 = last ? b1 : frag_b_bf16(wbase + 1 * 16 * C_ + k0 + 32, C_);
    v16bf bn2 = last ? b2 : frag_b_bf16(wbase + 2 * 16 * C_ + k0 + 32, C_);
    v16bf bn3 = last ? b3 : frag_b_bf16(wbase + 3 * 16 * C_ + k0 + 32, C_);
    acc[0] = __builtin_amdgcn_wmma_f32_16x16x32_bf16(false, a, false, b0, (short)0, acc[0], false, false);
    acc[1] = __builtin_amdgcn_wmma_f32_16x16x32_bf16(false, a, false, b1, (short)0, acc[1], false, false);
    acc[2] = __builtin_amdgcn_wmma_f32_16x16x32_bf16(false, a, false, b2, (short)0, acc[2], false, false);
    acc[3] = __builtin_amdgcn_wmma_f32_16x16x32_bf16(false, a, false, b3, (short)0, acc[3], false, false);
    a = an; b0 = bn0; b1 = bn1; b2 = bn2; b3 = bn3;
  }
  const int n = threadIdx.x & 15;
#pragma unroll
  for (int t = 0; t < 4; ++t) {
    float bv = bias[n0 + 16 * t + n];
    store_c(out + (size_t)m0 * C_ + n0 + 16 * t, C_, acc[t], bv);
  }
}

// ---- launch ----------------------------------------------------------------

extern "C" void kernel_launch(void* const* d_in, const int* in_sizes, int n_in,
                              void* d_out, int out_size, void* d_ws, size_t ws_size,
                              hipStream_t stream) {
  (void)in_sizes; (void)n_in; (void)out_size; (void)ws_size;
  const float* x     = (const float*)d_in[0];
  const float* Wqkv  = (const float*)d_in[1];
  const float* Wproj = (const float*)d_in[2];
  const float* bproj = (const float*)d_in[3];
  const float* ker   = (const float*)d_in[4];
  const float* kbias = (const float*)d_in[5];

  char* ws = (char*)d_ws;
  float*          qkv   = (float*)(ws);                               // 100,663,296 B
  float*          fa    = (float*)(ws + 100663296u);                  //  33,554,432 B
  unsigned short* xb    = (unsigned short*)(ws + 134217728u);         //  16,777,216 B
  unsigned short* outp  = (unsigned short*)(ws + 150994944u);         //  16,777,216 B
  unsigned short* wqb   = (unsigned short*)(ws + 167772160u);         //     393,216 B
  unsigned short* wpb   = (unsigned short*)(ws + 168165376u);         //     131,072 B
  float*          kvbuf = (float*)(ws + 168296448u);                  //     262,144 B
  float*          sume  = (float*)(ws + 168558592u);                  //       8,192 B

  hipLaunchKernelGGL(k_prep,      dim3(768),     dim3(256), 0, stream, Wqkv, Wproj, wqb, wpb, sume, kvbuf);
  hipLaunchKernelGGL(k_cvt_x,     dim3(8192),    dim3(256), 0, stream, x, xb);
  hipLaunchKernelGGL(k_gemm_qkv,  dim3(6, 512),  dim3(256), 0, stream, xb, wqb, qkv);
  hipLaunchKernelGGL(k_softsum,   dim3(128),     dim3(256), 0, stream, qkv, sume);
  hipLaunchKernelGGL(k_kv,        dim3(64, 8),   dim3(128), 0, stream, qkv, kvbuf);
  hipLaunchKernelGGL(k_fa,        dim3(64, 32),  dim3(256), 0, stream, qkv, kvbuf, sume, fa);
  hipLaunchKernelGGL(k_crpe,      dim3(32768),   dim3(256), 0, stream, qkv, fa, ker, kbias, outp);
  hipLaunchKernelGGL(k_gemm_proj, dim3(2, 512),  dim3(256), 0, stream, outp, wpb, bproj, (float*)d_out);
}
